// Attention_78374563217891
// MI455X (gfx1250) — matrix operand order repaired and checked by reading, not compiled, not run on hardware
//
#include <hip/hip_runtime.h>

typedef __attribute__((ext_vector_type(16))) _Float16 v16h;
typedef __attribute__((ext_vector_type(8)))  float    v8f;
typedef __attribute__((ext_vector_type(4)))  _Float16 v4h;

#define B_SZ   2
#define N_SEQ  2048
#define DIM_   1024
#define HEADS_ 16
#define DHEAD  64
#define ROWS   (B_SZ * N_SEQ)   /* 4096 */
#define CQKV   (3 * DIM_)       /* 3072 */

// ---------------- RMSNorm (fp32 in, f16 out) ----------------
__global__ __launch_bounds__(256) void k_rmsnorm(const float* __restrict__ x,
                                                 const float* __restrict__ gamma,
                                                 _Float16* __restrict__ xn) {
  __shared__ float red[8];
  __shared__ float rtot;
  const int row = blockIdx.x;
  const float4 v = ((const float4*)(x + (size_t)row * DIM_))[threadIdx.x];
  float ss = v.x * v.x + v.y * v.y + v.z * v.z + v.w * v.w;
  for (int d = 16; d > 0; d >>= 1) ss += __shfl_xor(ss, d, 32);
  if ((threadIdx.x & 31) == 0) red[threadIdx.x >> 5] = ss;
  __syncthreads();
  if (threadIdx.x == 0) {
    float t = 0.f;
    for (int i = 0; i < 8; ++i) t += red[i];
    rtot = rsqrtf(t * (1.0f / DIM_) + 1e-5f);
  }
  __syncthreads();
  const float r = rtot;
  const float4 g = ((const float4*)gamma)[threadIdx.x];
  v4h o;
  o[0] = (_Float16)(v.x * r * g.x); o[1] = (_Float16)(v.y * r * g.y);
  o[2] = (_Float16)(v.z * r * g.z); o[3] = (_Float16)(v.w * r * g.w);
  ((v4h*)(xn + (size_t)row * DIM_))[threadIdx.x] = o;
}

// ---------------- fp32 -> f16 weight convert ----------------
__global__ __launch_bounds__(256) void k_cvt_f16(const float* __restrict__ in,
                                                 _Float16* __restrict__ out, int n4) {
  int i = blockIdx.x * 256 + threadIdx.x;
  if (i < n4) {
    float4 v = ((const float4*)in)[i];
    v4h o;
    o[0] = (_Float16)v.x; o[1] = (_Float16)v.y;
    o[2] = (_Float16)v.z; o[3] = (_Float16)v.w;
    ((v4h*)out)[i] = o;
  }
}

// ---------------- 64x64 WMMA GEMM tile core (BK=32) ----------------
// A row-major [*, lda] f16; B row-major [K, ldb] f16 (staged transposed in LDS).
// 128 threads = 4 waves; wave w computes rows w*16..w*16+15, 4 N-tiles of 16.
__device__ __forceinline__ void gemm_tile_64x64(const _Float16* __restrict__ A, int lda,
                                                const _Float16* __restrict__ Bm, int ldb,
                                                int row0, int col0, int K,
                                                _Float16* lA, _Float16* lBt, v8f cacc[4]) {
  const int tid  = threadIdx.x;
  const int wave = tid >> 5;
  const int lane = tid & 31;
  const int nn   = lane & 15;
  const int kbA  = (lane < 16) ? 0 : 16;  // 16-bit A-matrix K halves
  const int kbB  = (lane < 16) ? 0 : 16;  // 16-bit B-matrix K halves
  v8f z = {0.f, 0.f, 0.f, 0.f, 0.f, 0.f, 0.f, 0.f};
  cacc[0] = z; cacc[1] = z; cacc[2] = z; cacc[3] = z;
  for (int kk = 0; kk < K; kk += 32) {
    // prefetch next K-step tiles into cache (global_prefetch_b8)
    if (kk + 32 < K) {
      __builtin_prefetch(A + (size_t)(row0 + (tid & 63)) * lda + kk + 32, 0, 3);
      __builtin_prefetch(Bm + (size_t)(kk + 32 + (tid & 31)) * ldb + col0, 0, 3);
    }
    __syncthreads();
    // A tile 64x32, row-major, 16B chunks
#pragma unroll
    for (int i = 0; i < 2; ++i) {
      int c = tid + i * 128;
      int r = c >> 2, co = (c & 3) * 8;
      *(float4*)(lA + r * 32 + co) =
          *(const float4*)(A + (size_t)(row0 + r) * lda + kk + co);
    }
    // B tile 32x64 -> transposed LDS lBt[n][k] so fragment reads are contiguous
#pragma unroll
    for (int i = 0; i < 2; ++i) {
      int c = tid + i * 128;
      int r = c >> 3, co = (c & 7) * 8;
      const _Float16* bp = Bm + (size_t)(kk + r) * ldb + col0 + co;
#pragma unroll
      for (int j = 0; j < 8; ++j) lBt[(co + j) * 32 + r] = bp[j];
    }
    __syncthreads();
    // A fragment: lane m = lane&15, lane half h holds contiguous k = 16h..16h+15
    v16h af;
    const _Float16* ap = lA + (wave * 16 + nn) * 32;
#pragma unroll
    for (int j = 0; j < 8; ++j) { af[j] = ap[kbA + j]; af[8 + j] = ap[kbA + 8 + j]; }
#pragma unroll
    for (int t = 0; t < 4; ++t) {
      v16h bf = *(const v16h*)(lBt + (t * 16 + nn) * 32 + kbB);
      cacc[t] = __builtin_amdgcn_wmma_f32_16x16x32_f16(false, af, false, bf,
                                                       (short)0, cacc[t], false, false);
    }
  }
}

// ---------------- QKV projection with scatter to [b,h,n,64] ----------------
__global__ __launch_bounds__(128) void k_gemm_qkv(const _Float16* __restrict__ xn,
                                                  const _Float16* __restrict__ w,
                                                  _Float16* __restrict__ Qo,
                                                  _Float16* __restrict__ Ko,
                                                  _Float16* __restrict__ Vo) {
  __shared__ __align__(32) _Float16 lA[64 * 32];
  __shared__ __align__(32) _Float16 lBt[64 * 32];
  const int row0 = blockIdx.y * 64;
  const int col0 = blockIdx.x * 64;
  v8f c[4];
  gemm_tile_64x64(xn, DIM_, w, CQKV, row0, col0, DIM_, lA, lBt, c);
  const int lane = threadIdx.x & 31;
  const int wave = threadIdx.x >> 5;
  const int nn   = lane & 15;
  const int half = (lane < 16) ? 0 : 8;
#pragma unroll
  for (int t = 0; t < 4; ++t) {
    int cg = col0 + t * 16 + nn;             // 0..3071 = s*1024 + h*64 + dc
    int s = cg >> 10, h = (cg >> 6) & 15, dc = cg & 63;
    _Float16* dst = (s == 0) ? Qo : ((s == 1) ? Ko : Vo);
#pragma unroll
    for (int r = 0; r < 8; ++r) {
      int rg = row0 + wave * 16 + r + half;
      int b = rg >> 11, nq = rg & (N_SEQ - 1);
      dst[((size_t)(b * HEADS_ + h) * N_SEQ + nq) * DHEAD + dc] = (_Float16)c[t][r];
    }
  }
}

// ---------------- Flash attention: one (b,h, 64-query) tile per block ----------------
__global__ __launch_bounds__(128) void k_attn(const _Float16* __restrict__ Q,
                                              const _Float16* __restrict__ K,
                                              const _Float16* __restrict__ V,
                                              const unsigned char* __restrict__ mask,
                                              _Float16* __restrict__ O) {
  __shared__ __align__(32) _Float16 lQ[64 * 64];
  __shared__ __align__(32) _Float16 lVt[64 * 32];
  __shared__ __align__(32) _Float16 lP[4][16 * 32];
  const int tid  = threadIdx.x, wave = tid >> 5, lane = tid & 31;
  const int nn   = lane & 15;
  const int half = (lane < 16) ? 0 : 8;
  const int kbA  = (lane < 16) ? 0 : 16;
  const int kbB  = (lane < 16) ? 0 : 16;
  const int bh = blockIdx.y;
  const int b  = bh >> 4;
  const int q0 = blockIdx.x * 64;
  const _Float16* Qb = Q + ((size_t)bh * N_SEQ + q0) * DHEAD;
  const _Float16* Kb = K + (size_t)bh * N_SEQ * DHEAD;
  const _Float16* Vb = V + (size_t)bh * N_SEQ * DHEAD;
  const unsigned char* mb = mask + b * N_SEQ;

  // stage 64x64 Q tile
#pragma unroll
  for (int i = 0; i < 4; ++i) {
    int c = tid + i * 128;
    int r = c >> 3, co = (c & 7) * 8;
    *(float4*)(lQ + r * 64 + co) = *(const float4*)(Qb + r * 64 + co);
  }
  __syncthreads();
  // pre-pack Q A-fragments (2 k-steps over dh=64)
  v16h qf[2];
#pragma unroll
  for (int s = 0; s < 2; ++s) {
    const _Float16* qp = lQ + (wave * 16 + nn) * 64 + s * 32;
#pragma unroll
    for (int j = 0; j < 8; ++j) { qf[s][j] = qp[kbA + j]; qf[s][8 + j] = qp[kbA + 8 + j]; }
  }

  v8f z = {0.f, 0.f, 0.f, 0.f, 0.f, 0.f, 0.f, 0.f};
  v8f o[4]; o[0] = z; o[1] = z; o[2] = z; o[3] = z;
  float mrun[8], lrun[8];
#pragma unroll
  for (int r = 0; r < 8; ++r) { mrun[r] = -1e30f; lrun[r] = 0.f; }

  for (int kt = 0; kt < N_SEQ; kt += 32) {
    // prefetch next key/value tiles (32 rows x 128B each) while we compute
    if (kt + 32 < N_SEQ) {
      __builtin_prefetch(Kb + (size_t)(kt + 32 + (tid & 31)) * DHEAD, 0, 3);
      __builtin_prefetch(Vb + (size_t)(kt + 32 + (tid & 31)) * DHEAD, 0, 3);
    }
    __syncthreads();  // guard lVt reuse
    // stage V tile (32 keys x 64) transposed: lVt[dc][key]
#pragma unroll
    for (int i = 0; i < 2; ++i) {
      int c = tid + i * 128;
      int key = c >> 3, co = (c & 7) * 8;
      const _Float16* vp = Vb + (size_t)(kt + key) * DHEAD + co;
#pragma unroll
      for (int j = 0; j < 8; ++j) lVt[(co + j) * 32 + key] = vp[j];
    }
    __syncthreads();

    // S = Q K^T (16 rows x 32 keys per wave); K fragments straight from global
    v8f s0 = z, s1 = z;
#pragma unroll
    for (int ks = 0; ks < 2; ++ks) {
      v16h kf0 = *(const v16h*)(Kb + (size_t)(kt + nn) * DHEAD + ks * 32 + kbB);
      v16h kf1 = *(const v16h*)(Kb + (size_t)(kt + 16 + nn) * DHEAD + ks * 32 + kbB);
      s0 = __builtin_amdgcn_wmma_f32_16x16x32_f16(false, qf[ks], false, kf0, (short)0, s0, false, false);
      s1 = __builtin_amdgcn_wmma_f32_16x16x32_f16(false, qf[ks], false, kf1, (short)0, s1, false, false);
    }
    const bool ok0 = mb[kt + nn] != 0;
    const bool ok1 = mb[kt + 16 + nn] != 0;
#pragma unroll
    for (int r = 0; r < 8; ++r) {
      float a0 = s0[r] * 8.0f;   // reference multiplies q by sqrt(dh)=8
      float a1 = s1[r] * 8.0f;
      if (!ok0) a0 = -3.0e38f;
      if (!ok1) a1 = -3.0e38f;
      s0[r] = a0; s1[r] = a1;
    }
    // online softmax: rows live across 16 lanes of the C layout
    float alpha[8];
#pragma unroll
    for (int r = 0; r < 8; ++r) {
      float v = fmaxf(s0[r], s1[r]);
      for (int d = 1; d < 16; d <<= 1) v = fmaxf(v, __shfl_xor(v, d, 32));
      float mnew = fmaxf(mrun[r], v);
      alpha[r] = __expf(mrun[r] - mnew);
      float p0 = __expf(s0[r] - mnew);
      float p1 = __expf(s1[r] - mnew);
      s0[r] = p0; s1[r] = p1;
      float rs = p0 + p1;
      for (int d = 1; d < 16; d <<= 1) rs += __shfl_xor(rs, d, 32);
      lrun[r] = lrun[r] * alpha[r] + rs;
      mrun[r] = mnew;
    }
#pragma unroll
    for (int t = 0; t < 4; ++t)
#pragma unroll
      for (int r = 0; r < 8; ++r) o[t][r] *= alpha[r];
    // C-layout P -> per-wave LDS -> A-layout fragment (f16)
    _Float16* lpw = lP[wave];
#pragma unroll
    for (int r = 0; r < 8; ++r) {
      lpw[(r + half) * 32 + nn]      = (_Float16)s0[r];
      lpw[(r + half) * 32 + 16 + nn] = (_Float16)s1[r];
    }
    v16h pf;
    const _Float16* pp = lpw + nn * 32;
#pragma unroll
    for (int j = 0; j < 8; ++j) { pf[j] = pp[kbA + j]; pf[8 + j] = pp[kbA + 8 + j]; }
    // O += P @ Vtile (4 dc-tiles of 16)
#pragma unroll
    for (int t = 0; t < 4; ++t) {
      v16h vf = *(const v16h*)(lVt + (t * 16 + nn) * 32 + kbB);
      o[t] = __builtin_amdgcn_wmma_f32_16x16x32_f16(false, pf, false, vf, (short)0, o[t], false, false);
    }
  }
  // normalize and write [b, n, h*64+dc] as f16
  const int h = bh & 15;
#pragma unroll
  for (int t = 0; t < 4; ++t)
#pragma unroll
    for (int r = 0; r < 8; ++r) {
      int rowq = q0 + wave * 16 + r + half;
      float val = o[t][r] / lrun[r];
      O[((size_t)b * N_SEQ + rowq) * DIM_ + h * DHEAD + t * 16 + nn] = (_Float16)val;
    }
}

// ---------------- Output projection (fp32 out) ----------------
__global__ __launch_bounds__(128) void k_gemm_out(const _Float16* __restrict__ Ain,
                                                  const _Float16* __restrict__ w,
                                                  float* __restrict__ out) {
  __shared__ __align__(32) _Float16 lA[64 * 32];
  __shared__ __align__(32) _Float16 lBt[64 * 32];
  const int row0 = blockIdx.y * 64;
  const int col0 = blockIdx.x * 64;
  v8f c[4];
  gemm_tile_64x64(Ain, DIM_, w, DIM_, row0, col0, DIM_, lA, lBt, c);
  const int lane = threadIdx.x & 31;
  const int wave = threadIdx.x >> 5;
  const int nn   = lane & 15;
  const int half = (lane < 16) ? 0 : 8;
#pragma unroll
  for (int t = 0; t < 4; ++t) {
    int cg = col0 + t * 16 + nn;
#pragma unroll
    for (int r = 0; r < 8; ++r) {
      int rg = row0 + wave * 16 + r + half;
      out[(size_t)rg * DIM_ + cg] = c[t][r];
    }
  }
}

extern "C" void kernel_launch(void* const* d_in, const int* in_sizes, int n_in,
                              void* d_out, int out_size, void* d_ws, size_t ws_size,
                              hipStream_t stream) {
  (void)in_sizes; (void)n_in; (void)out_size; (void)ws_size;
  const float*         x     = (const float*)d_in[0];
  const unsigned char* mask  = (const unsigned char*)d_in[1];
  const float*         gamma = (const float*)d_in[2];
  const float*         wqkv  = (const float*)d_in[3];
  const float*         wout  = (const float*)d_in[4];
  float* out = (float*)d_out;

  char* ws = (char*)d_ws;
  size_t off = 0;
  _Float16* xn   = (_Float16*)(ws + off); off += (size_t)ROWS * DIM_ * 2;   // also attn-out
  _Float16* wq16 = (_Float16*)(ws + off); off += (size_t)DIM_ * CQKV * 2;
  _Float16* wo16 = (_Float16*)(ws + off); off += (size_t)DIM_ * DIM_ * 2;
  _Float16* Qb   = (_Float16*)(ws + off); off += (size_t)ROWS * DIM_ * 2;
  _Float16* Kb   = (_Float16*)(ws + off); off += (size_t)ROWS * DIM_ * 2;
  _Float16* Vb   = (_Float16*)(ws + off); off += (size_t)ROWS * DIM_ * 2;

  k_rmsnorm<<<ROWS, 256, 0, stream>>>(x, gamma, xn);
  k_cvt_f16<<<(DIM_ * CQKV / 4 + 255) / 256, 256, 0, stream>>>(wqkv, wq16, DIM_ * CQKV / 4);
  k_cvt_f16<<<(DIM_ * DIM_ / 4 + 255) / 256, 256, 0, stream>>>(wout, wo16, DIM_ * DIM_ / 4);
  k_gemm_qkv<<<dim3(CQKV / 64, ROWS / 64), 128, 0, stream>>>(xn, wq16, Qb, Kb, Vb);
  k_attn<<<dim3(N_SEQ / 64, B_SZ * HEADS_), 128, 0, stream>>>(Qb, Kb, Vb, mask, xn);
  k_gemm_out<<<dim3(DIM_ / 64, ROWS / 64), 128, 0, stream>>>(xn, wo16, out);
}
